// GATConvNetwork_86088324481811
// MI455X (gfx1250) — matrix operand order repaired and checked by reading, not compiled, not run on hardware
//
#include <hip/hip_runtime.h>
#include <hip/hip_bf16.h>

typedef _Float16 v8h  __attribute__((ext_vector_type(8)));
typedef _Float16 v16h __attribute__((ext_vector_type(16)));
typedef float    v8f  __attribute__((ext_vector_type(8)));

#define HC 512
#define CH 256

__device__ __forceinline__ float lrelu(float x, float s) { return x > 0.0f ? x : s * x; }
__device__ __forceinline__ unsigned orderf(float x) {
    unsigned u = __float_as_uint(x);
    return (u & 0x80000000u) ? ~u : (u | 0x80000000u);
}
__device__ __forceinline__ float deorderf(unsigned u) {
    return __uint_as_float((u & 0x80000000u) ? (u ^ 0x80000000u) : ~u);
}

// ---------------- input MLPs ----------------
__global__ void __launch_bounds__(128)
node_in_kernel(const float* __restrict__ x,
               const float* __restrict__ W1, const float* __restrict__ b1,
               const float* __restrict__ W2, const float* __restrict__ b2,
               const float* __restrict__ W3, const float* __restrict__ b3,
               float* __restrict__ v0, int Nn) {
    int n = blockIdx.x * blockDim.x + threadIdx.x;
    if (n >= Nn) return;
    float xin[9];
#pragma unroll
    for (int i = 0; i < 9; ++i) xin[i] = x[n * 9 + i];
    float h1[32];
#pragma unroll
    for (int j = 0; j < 32; ++j) {
        float a = b1[j];
#pragma unroll
        for (int k = 0; k < 9; ++k) a += xin[k] * W1[k * 32 + j];
        h1[j] = fmaxf(a, 0.0f);
    }
    float h2[64];
#pragma unroll
    for (int j = 0; j < 64; ++j) {
        float a = b2[j];
#pragma unroll
        for (int k = 0; k < 32; ++k) a += h1[k] * W2[k * 64 + j];
        h2[j] = fmaxf(a, 0.0f);
    }
    for (int j = 0; j < CH; ++j) {
        float a = b3[j];
#pragma unroll
        for (int k = 0; k < 64; ++k) a += h2[k] * W3[k * CH + j];
        v0[(size_t)n * CH + j] = fmaxf(a, 0.0f);
    }
}

__global__ void __launch_bounds__(128)
edge_in_kernel(const float* __restrict__ ea,
               const float* __restrict__ W1, const float* __restrict__ b1,
               const float* __restrict__ W2, const float* __restrict__ b2,
               const float* __restrict__ W3, const float* __restrict__ b3,
               float* __restrict__ e_val, int Ne) {
    int e = blockIdx.x * blockDim.x + threadIdx.x;
    if (e >= Ne) return;
    float xin[4];
#pragma unroll
    for (int i = 0; i < 4; ++i) xin[i] = ea[e * 4 + i];
    float h1[32];
#pragma unroll
    for (int j = 0; j < 32; ++j) {
        float a = b1[j];
#pragma unroll
        for (int k = 0; k < 4; ++k) a += xin[k] * W1[k * 32 + j];
        h1[j] = fmaxf(a, 0.0f);
    }
    float out = b3[0];
#pragma unroll
    for (int j = 0; j < 64; ++j) {
        float a = b2[j];
#pragma unroll
        for (int k = 0; k < 32; ++k) a += h1[k] * W2[k * 64 + j];
        out += fmaxf(a, 0.0f) * W3[j];
    }
    e_val[e] = fmaxf(out, 0.0f);
}

// ---------------- WMMA GEMM: C[M x 512] = A[M x K] * B[K x 512] ----------------
// Block: 128 threads = 4 waves in a 2x2 grid of 16x16 WMMA tiles -> 32x32 block tile.
// Double-buffered LDS staging: tile t+1's global_load_b128s issue while tile t's
// v_wmma executes; one barrier per K step.
__global__ void __launch_bounds__(128)
wmma_gemm_kernel(const float* __restrict__ A, const float* __restrict__ B,
                 float* __restrict__ C, int M, int K) {
    __shared__ alignas(16) _Float16 As[2][32 * 32];     // row-major [m][k]
    __shared__ alignas(16) _Float16 Bs[2][32 * 32];     // transposed [n][k]
    const int tid  = threadIdx.x;
    const int lane = tid & 31;
    const int wave = tid >> 5;
    const int m0 = blockIdx.x * 32;
    const int n0 = blockIdx.y * 32;
    const int wm = (wave & 1) * 16;
    const int wn = (wave >> 1) * 16;
    const int lrow = lane & 15;
    const int hi   = lane >> 4;     // 0 for lanes 0-15, 1 for lanes 16-31

    // staging coordinates: thread covers 8 contiguous elements of one row
    const int ar = tid >> 2;            // row within tile (A: m, B: k)
    const int ac = (tid & 3) * 8;       // 8-wide column segment
    const int gm = m0 + ar;
    const int gmc = (gm < M) ? gm : (M - 1);        // clamp, branchless tail
    const float rowScale = (gm < M) ? 1.0f : 0.0f;  // zero-fill tail rows
    const float* __restrict__ aRow = A + (size_t)gmc * K + ac;
    const float* __restrict__ bRow = B + (size_t)ar * HC + n0 + ac;

    auto stage = [&](int buf, int k0) {
        float4 a0 = *(const float4*)(aRow + k0);
        float4 a1 = *(const float4*)(aRow + k0 + 4);
        float4 b0 = *(const float4*)(bRow + (size_t)k0 * HC);
        float4 b1 = *(const float4*)(bRow + (size_t)k0 * HC + 4);
        v8h av;
        av[0] = (_Float16)(a0.x * rowScale);
        av[1] = (_Float16)(a0.y * rowScale);
        av[2] = (_Float16)(a0.z * rowScale);
        av[3] = (_Float16)(a0.w * rowScale);
        av[4] = (_Float16)(a1.x * rowScale);
        av[5] = (_Float16)(a1.y * rowScale);
        av[6] = (_Float16)(a1.z * rowScale);
        av[7] = (_Float16)(a1.w * rowScale);
        *(v8h*)&As[buf][ar * 32 + ac] = av;             // one 16B LDS store
        _Float16 bh[8];
        bh[0] = (_Float16)b0.x; bh[1] = (_Float16)b0.y;
        bh[2] = (_Float16)b0.z; bh[3] = (_Float16)b0.w;
        bh[4] = (_Float16)b1.x; bh[5] = (_Float16)b1.y;
        bh[6] = (_Float16)b1.z; bh[7] = (_Float16)b1.w;
#pragma unroll
        for (int i = 0; i < 8; ++i)                     // transposed scatter
            Bs[buf][(ac + i) * 32 + ar] = bh[i];
    };

    stage(0, 0);
    v8f acc = {};
    const int ksteps = K / 32;
    for (int kt = 0; kt < ksteps; ++kt) {
        const int buf = kt & 1;
        __syncthreads();
        if (kt + 1 < ksteps) stage(buf ^ 1, (kt + 1) * 32);
        // A fragment: lanes 0-15 -> K[0..7],[16..23]; lanes 16-31 -> K[8..15],[24..31]
        v16h af, bf;
        {
            const _Float16* ap = &As[buf][(wm + lrow) * 32 + hi * 8];
            v8h lo  = *(const v8h*)(ap);
            v8h hi8 = *(const v8h*)(ap + 16);
#pragma unroll
            for (int i = 0; i < 8; ++i) { af[i] = lo[i]; af[i + 8] = hi8[i]; }
        }
        // B fragment (transposed LDS): lanes 0-15 -> K[0..7],[16..23]; lanes 16-31 -> K[8..15],[24..31]
        {
            const _Float16* bp = &Bs[buf][(wn + lrow) * 32 + hi * 8];
            v8h lo  = *(const v8h*)(bp);
            v8h hi8 = *(const v8h*)(bp + 16);
#pragma unroll
            for (int i = 0; i < 8; ++i) { bf[i] = lo[i]; bf[i + 8] = hi8[i]; }
        }
        acc = __builtin_amdgcn_wmma_f32_16x16x32_f16(false, af, false, bf,
                                                     (short)0, acc, false, false);
    }
    const int col = n0 + wn + lrow;
#pragma unroll
    for (int r = 0; r < 8; ++r) {
        int row = m0 + wm + r + hi * 8;
        if (row < M) C[(size_t)row * HC + col] = acc[r];
    }
}

// ---------------- per-node attention dot products ----------------
__global__ void __launch_bounds__(64)
att_reduce_kernel(const float* __restrict__ h, const float* __restrict__ att_src,
                  const float* __restrict__ att_dst, float* __restrict__ asrc,
                  float* __restrict__ adst) {
    int n = blockIdx.x;
    int head = threadIdx.x >> 5;
    int lane = threadIdx.x & 31;
    const float* hp = h + (size_t)n * HC + head * CH;
    const float* as = att_src + head * CH;
    const float* ad = att_dst + head * CH;
    float s1 = 0.f, s2 = 0.f;
#pragma unroll
    for (int i = 0; i < 8; ++i) {
        int c = lane + i * 32;
        float v = hp[c];
        s1 += v * as[c];
        s2 += v * ad[c];
    }
#pragma unroll
    for (int m = 16; m > 0; m >>= 1) {
        s1 += __shfl_xor(s1, m, 32);
        s2 += __shfl_xor(s2, m, 32);
    }
    if (lane == 0) { asrc[n * 2 + head] = s1; adst[n * 2 + head] = s2; }
}

__global__ void __launch_bounds__(64)
we_att_kernel(const float* __restrict__ We, const float* __restrict__ att_edge,
              float* __restrict__ we_att) {
    int head = threadIdx.x >> 5;
    int lane = threadIdx.x & 31;
    float s = 0.f;
#pragma unroll
    for (int i = 0; i < 8; ++i) {
        int c = head * CH + lane + i * 32;
        s += We[c] * att_edge[c];
    }
#pragma unroll
    for (int m = 16; m > 0; m >>= 1) s += __shfl_xor(s, m, 32);
    if (lane == 0) we_att[head] = s;
}

// ---------------- per-layer init ----------------
__global__ void __launch_bounds__(256)
init_layer_kernel(float* __restrict__ accum, float* __restrict__ den,
                  unsigned* __restrict__ amax, int Nn) {
    int i = blockIdx.x * blockDim.x + threadIdx.x;
    if (i < Nn * HC) accum[i] = 0.0f;
    if (i < Nn * 2) { den[i] = 0.0f; amax[i] = 0x007FFFFFu; /* ordered(-inf) */ }
}

// ---------------- edge softmax ----------------
__global__ void __launch_bounds__(256)
edge_alpha_kernel(const int* __restrict__ src, const int* __restrict__ dst,
                  const float* __restrict__ asrc, const float* __restrict__ adst,
                  const float* __restrict__ e_val, const float* __restrict__ we_att,
                  float* __restrict__ exbuf, unsigned* __restrict__ amax, int Ne) {
    int e = blockIdx.x * blockDim.x + threadIdx.x;
    if (e >= Ne) return;
    int s = src[e], d = dst[e];
    float ev = e_val[e];
#pragma unroll
    for (int h = 0; h < 2; ++h) {
        float al = asrc[s * 2 + h] + adst[d * 2 + h] + ev * we_att[h];
        al = lrelu(al, 0.2f);
        exbuf[e * 2 + h] = al;
        atomicMax(&amax[d * 2 + h], orderf(al));
    }
}

__global__ void __launch_bounds__(256)
edge_exp_kernel(const int* __restrict__ dst, float* __restrict__ exbuf,
                const unsigned* __restrict__ amax, float* __restrict__ den, int Ne) {
    int e = blockIdx.x * blockDim.x + threadIdx.x;
    if (e >= Ne) return;
    int d = dst[e];
#pragma unroll
    for (int h = 0; h < 2; ++h) {
        float mx = deorderf(amax[d * 2 + h]);
        float ex = __expf(exbuf[e * 2 + h] - mx);
        exbuf[e * 2 + h] = ex;
        unsafeAtomicAdd(&den[d * 2 + h], ex);
    }
}

// ---------------- message passing (memory-bound hot loop) ----------------
__global__ void __launch_bounds__(128)
edge_message_kernel(const int* __restrict__ src, const int* __restrict__ dst,
                    const float* __restrict__ exbuf, const float* __restrict__ den,
                    const float* __restrict__ hmat, float* __restrict__ accum,
                    float* __restrict__ abuf) {
    int e = blockIdx.x;
    int tid = threadIdx.x;
    int s = src[e], d = dst[e];
    int head = tid >> 6;
    float a = exbuf[e * 2 + head] / (den[d * 2 + head] + 1e-16f);
    if ((tid & 63) == 0) abuf[e * 2 + head] = a;
    int c = tid * 4;
    const float4 hv = *(const float4*)(hmat + (size_t)s * HC + c);
    float* out = accum + (size_t)d * HC + c;
    unsafeAtomicAdd(out + 0, a * hv.x);
    unsafeAtomicAdd(out + 1, a * hv.y);
    unsafeAtomicAdd(out + 2, a * hv.z);
    unsafeAtomicAdd(out + 3, a * hv.w);
}

// ---------------- node update: +bias -> layernorm -> leaky_relu 0.01 ----------------
__global__ void __launch_bounds__(256)
node_update_kernel(const float* __restrict__ accum, const float* __restrict__ bias,
                   const float* __restrict__ g, const float* __restrict__ b,
                   float* __restrict__ vout, int Nn) {
    int wave = threadIdx.x >> 5, lane = threadIdx.x & 31;
    int n = blockIdx.x * 8 + wave;
    if (n >= Nn) return;
    const float* ap = accum + (size_t)n * HC;
    float x[16];
    float s = 0.f, sq = 0.f;
#pragma unroll
    for (int i = 0; i < 16; ++i) {
        int c = lane + i * 32;
        float v = ap[c] + bias[c];
        x[i] = v; s += v; sq += v * v;
    }
#pragma unroll
    for (int m = 16; m > 0; m >>= 1) {
        s += __shfl_xor(s, m, 32);
        sq += __shfl_xor(sq, m, 32);
    }
    float mean = s * (1.0f / HC);
    float var  = sq * (1.0f / HC) - mean * mean;
    float rstd = rsqrtf(var + 1e-5f);
    float* vp = vout + (size_t)n * HC;
#pragma unroll
    for (int i = 0; i < 16; ++i) {
        int c = lane + i * 32;
        float v = (x[i] - mean) * rstd * g[c] + b[c];
        vp[c] = lrelu(v, 0.01f);
    }
}

// ---------------- edge output head of GAT layer ----------------
__global__ void __launch_bounds__(256)
edge_out_kernel(const float* __restrict__ abuf, const float* __restrict__ eg,
                const float* __restrict__ eb, const float* __restrict__ elW,
                const float* __restrict__ elb, float* __restrict__ e_val, int Ne) {
    int e = blockIdx.x * blockDim.x + threadIdx.x;
    if (e >= Ne) return;
    float a0 = abuf[e * 2], a1 = abuf[e * 2 + 1];
    float m = 0.5f * (a0 + a1);
    float d0 = a0 - m, d1 = a1 - m;
    float rs = rsqrtf(0.5f * (d0 * d0 + d1 * d1) + 1e-5f);
    float n0 = d0 * rs * eg[0] + eb[0];
    float n1 = d1 * rs * eg[1] + eb[1];
    float o = n0 * elW[0] + n1 * elW[1] + elb[0];
    e_val[e] = fmaxf(o, 0.0f);
}

// ---------------- output MLPs (no relu) ----------------
__global__ void __launch_bounds__(64)
node_out_kernel(const float* __restrict__ v,
                const float* __restrict__ W1, const float* __restrict__ b1,
                const float* __restrict__ W2, const float* __restrict__ b2,
                const float* __restrict__ W3, const float* __restrict__ b3,
                float* __restrict__ out) {
    __shared__ float vs[HC];
    __shared__ float h1[64];
    __shared__ float h2[32];
    int n = blockIdx.x;
    int tid = threadIdx.x;
    const float* vp = v + (size_t)n * HC;
    for (int i = tid; i < HC; i += 64) vs[i] = vp[i];
    __syncthreads();
    float acc = b1[tid];
    for (int k = 0; k < HC; ++k) acc += vs[k] * W1[k * 64 + tid];
    h1[tid] = acc;
    __syncthreads();
    if (tid < 32) {
        float a2 = b2[tid];
#pragma unroll
        for (int k = 0; k < 64; ++k) a2 += h1[k] * W2[k * 32 + tid];
        h2[tid] = a2;
    }
    __syncthreads();
    if (tid == 0) {
        float o = b3[0];
#pragma unroll
        for (int k = 0; k < 32; ++k) o += h2[k] * W3[k];
        out[n] = o;
    }
}

__global__ void __launch_bounds__(256)
edge_final_kernel(const float* __restrict__ e_val,
                  const float* __restrict__ W1, const float* __restrict__ b1,
                  const float* __restrict__ W2, const float* __restrict__ b2,
                  const float* __restrict__ W3, const float* __restrict__ b3,
                  float* __restrict__ out, int Ne) {
    int e = blockIdx.x * blockDim.x + threadIdx.x;
    if (e >= Ne) return;
    float ev = e_val[e];
    float h2[32];
#pragma unroll
    for (int j = 0; j < 32; ++j) h2[j] = b2[j];
    for (int k = 0; k < 64; ++k) {
        float h1 = ev * W1[k] + b1[k];
#pragma unroll
        for (int j = 0; j < 32; ++j) h2[j] += h1 * W2[k * 32 + j];
    }
    float o0 = b3[0], o1 = b3[1];
#pragma unroll
    for (int j = 0; j < 32; ++j) { o0 += h2[j] * W3[j * 2]; o1 += h2[j] * W3[j * 2 + 1]; }
    out[e * 2 + 0] = o0;
    out[e * 2 + 1] = o1;
}

// ---------------- host-side orchestration ----------------
extern "C" void kernel_launch(void* const* d_in, const int* in_sizes, int n_in,
                              void* d_out, int out_size, void* d_ws, size_t ws_size,
                              hipStream_t stream) {
    const int N = in_sizes[0] / 9;   // nodes
    const int E = in_sizes[1] / 4;   // edges

    const float* x  = (const float*)d_in[0];
    const float* ea = (const float*)d_in[1];
    const int*   ei = (const int*)d_in[2];
    const int* srcI = ei;
    const int* dstI = ei + E;

    // params flattened depth-first in dict order starting at index 3
    const float* const* P = (const float* const*)(d_in + 3);
    // in_v: 0..5, in_e: 6..11, gat[l]: 12 + 12*l .. +11, out_n: 60..65, out_e: 66..71

    // workspace layout (floats)
    float* ws = (float*)d_ws;
    size_t o = 0;
    float*    vbuf  = ws + o; o += (size_t)N * HC;
    float*    hbuf  = ws + o; o += (size_t)N * HC;
    float*    accum = ws + o; o += (size_t)N * HC;
    float*    asrc  = ws + o; o += (size_t)N * 2;
    float*    adst  = ws + o; o += (size_t)N * 2;
    float*    den   = ws + o; o += (size_t)N * 2;
    unsigned* amax  = (unsigned*)(ws + o); o += (size_t)N * 2;
    float*    exbuf = ws + o; o += (size_t)E * 2;
    float*    abuf  = ws + o; o += (size_t)E * 2;
    float*    e_val = ws + o; o += (size_t)E;
    float*    we_at = ws + o; o += 2;
    (void)ws_size; (void)n_in; (void)out_size;

    float* outv = (float*)d_out;
    float* oute = outv + N;

    // input MLPs
    node_in_kernel<<<(N + 127) / 128, 128, 0, stream>>>(
        x, P[0], P[1], P[2], P[3], P[4], P[5], vbuf, N);
    edge_in_kernel<<<(E + 127) / 128, 128, 0, stream>>>(
        ea, P[6], P[7], P[8], P[9], P[10], P[11], e_val, E);

    for (int layer = 0; layer < 4; ++layer) {
        const float* const* G = P + 12 + 12 * layer;
        // G: 0=W 1=We 2=att_src 3=att_dst 4=att_edge 5=bias 6=ln_g 7=ln_b
        //    8=eln_g 9=eln_b 10=el_W 11=el_b
        const int K = (layer == 0) ? CH : HC;

        dim3 ggrid((N + 31) / 32, HC / 32);
        wmma_gemm_kernel<<<ggrid, 128, 0, stream>>>(vbuf, G[0], hbuf, N, K);

        att_reduce_kernel<<<N, 64, 0, stream>>>(hbuf, G[2], G[3], asrc, adst);
        we_att_kernel<<<1, 64, 0, stream>>>(G[1], G[4], we_at);
        init_layer_kernel<<<((size_t)N * HC + 255) / 256, 256, 0, stream>>>(accum, den, amax, N);
        edge_alpha_kernel<<<(E + 255) / 256, 256, 0, stream>>>(
            srcI, dstI, asrc, adst, e_val, we_at, exbuf, amax, E);
        edge_exp_kernel<<<(E + 255) / 256, 256, 0, stream>>>(dstI, exbuf, amax, den, E);
        edge_message_kernel<<<E, 128, 0, stream>>>(
            srcI, dstI, exbuf, den, hbuf, accum, abuf);
        node_update_kernel<<<(N + 7) / 8, 256, 0, stream>>>(
            accum, G[5], G[6], G[7], vbuf, N);
        edge_out_kernel<<<(E + 255) / 256, 256, 0, stream>>>(
            abuf, G[8], G[9], G[10], G[11], e_val, E);
    }

    // output heads
    node_out_kernel<<<N, 64, 0, stream>>>(
        vbuf, P[60], P[61], P[62], P[63], P[64], P[65], outv);
    edge_final_kernel<<<(E + 255) / 256, 256, 0, stream>>>(
        e_val, P[66], P[67], P[68], P[69], P[70], P[71], oute, E);
}
